// Attention_69063074119685
// MI455X (gfx1250) — matrix operand order repaired and checked by reading, not compiled, not run on hardware
//
#include <hip/hip_runtime.h>
#include <hip/hip_bf16.h>
#include <math.h>

typedef unsigned short     u16;
typedef unsigned int       u32;
typedef unsigned long long u64;
typedef __attribute__((ext_vector_type(16))) __bf16 v16bf;
typedef __attribute__((ext_vector_type(8)))  float  v8f;
typedef __attribute__((ext_vector_type(4)))  u32    v4u;

#define DIM_   2048
#define S_LEN  1024
#define BATCH  8
#define NH     16
#define NKV    4
#define HD     128
#define MROWS  (BATCH * S_LEN)   /* 8192 */
#define KVDIM  (NKV * HD)        /* 512  */

#define BM 128
#define BN 128
#define BK 32

// ---------------- helpers ----------------

__device__ inline u16 f2bf(float f) {
  u32 u = __float_as_uint(f);
  u32 r = (u + 0x7FFFu + ((u >> 16) & 1u)) >> 16;  // round-to-nearest-even
  return (u16)r;
}

union FragCvt { v4u u[2]; v16bf v; };

// A-matrix 16x32 bf16 fragment (ISA 7.12.2): lane r=lane&15 is row M,
// half=lane>>4 selects K-base 0/8; per-lane K = {kb..kb+7, kb+16..kb+23}.
__device__ inline v16bf load_frag_a(const u16* p, int row0, int k0, int ld) {
  int lane = threadIdx.x & 31;
  int r = lane & 15, h = lane >> 4;
  const u16* q = p + (size_t)(row0 + r) * ld + k0 + h * 8;
  FragCvt c;
  c.u[0] = *(const v4u*)q;
  c.u[1] = *(const v4u*)(q + 16);
  return c.v;
}

// B-matrix 32x16 bf16 fragment: lane r is column N, per-lane K = {kb..kb+7, kb+16..kb+23}
// with kb = half*8 (same as A). Source is row-major W[n][k].
__device__ inline v16bf load_frag_b(const u16* p, int col0, int k0, int ld) {
  int lane = threadIdx.x & 31;
  int r = lane & 15, h = lane >> 4;
  const u16* q = p + (size_t)(col0 + r) * ld + k0 + h * 8;
  FragCvt c;
  c.u[0] = *(const v4u*)q;
  c.u[1] = *(const v4u*)(q + 16);
  return c.v;
}

__device__ inline v8f wmma_bf16(v16bf a, v16bf b, v8f c) {
  return __builtin_amdgcn_wmma_f32_16x16x32_bf16(
      /*neg_a=*/false, a, /*neg_b=*/false, b,
      /*c_mod=*/(short)0, c, /*reuse_a=*/false, /*reuse_b=*/false);
}

// Async global->LDS staging of a [rows x 32] bf16 tile (row-major, ld=32 in LDS).
// 256 threads move tile rows row0..row0+127 of a row-major [ldg] matrix.
// Each thread issues two GLOBAL_LOAD_ASYNC_TO_LDS_B128 (ASYNCcnt-tracked).
__device__ inline void stage_async(const u16* __restrict__ gbase, int ldg,
                                   int row0, int k0, u16* lds, int tid) {
#pragma unroll
  for (int i = 0; i < 2; ++i) {
    int c = tid * 2 + i;            // 0..511 : 128 rows x 4 16B segments
    int row = c >> 2, seg = c & 3;
    u64 g = (u64)(uintptr_t)(gbase + (size_t)(row0 + row) * ldg + k0 + seg * 8);
    u32 l = (u32)(uintptr_t)(lds + row * BK + seg * 8);
    asm volatile("global_load_async_to_lds_b128 %0, %1, off"
                 :: "v"(l), "v"(g) : "memory");
  }
}

__device__ inline void wait_async_all() {
  asm volatile("s_wait_asynccnt 0" ::: "memory");
}

// ---------------- kernels ----------------

__global__ void k_f32_to_bf16(const float* __restrict__ in,
                              u16* __restrict__ out, long long n) {
  long long i = (long long)blockIdx.x * blockDim.x + threadIdx.x;
  if (i < n) out[i] = f2bf(in[i]);
}

// RoPE: read fp32 rows, rotate pairs, write bf16 (same layout).
__global__ void k_rope_bf16(const float* __restrict__ in,
                            const float* __restrict__ cosT,
                            const float* __restrict__ sinT,
                            u16* __restrict__ out, int nheads) {
  long long pid = (long long)blockIdx.x * blockDim.x + threadIdx.x;
  long long total = (long long)MROWS * nheads * (HD / 2);
  if (pid >= total) return;
  int i = (int)(pid % (HD / 2));
  int h = (int)((pid / (HD / 2)) % nheads);
  long long row = pid / ((long long)(HD / 2) * nheads);
  int s = (int)(row % S_LEN);
  int ld = nheads * HD;
  const float* pr = in + row * ld + h * HD + 2 * i;
  float tr = pr[0], ti = pr[1];
  float c  = cosT[s * (HD / 2) + i];
  float sn = sinT[s * (HD / 2) + i];
  u16* po = out + row * ld + h * HD + 2 * i;
  po[0] = f2bf(tr * c - ti * sn);
  po[1] = f2bf(tr * sn + ti * c);
}

// V fp32 [B*S][512] -> Vt bf16 [B][G][HD][S]
__global__ void k_transpose_v(const float* __restrict__ v, u16* __restrict__ vt) {
  long long pid = (long long)blockIdx.x * blockDim.x + threadIdx.x;
  long long total = (long long)BATCH * NKV * HD * S_LEN;
  if (pid >= total) return;
  int s  = (int)(pid % S_LEN);
  int d  = (int)((pid / S_LEN) % HD);
  int bg = (int)(pid / ((long long)S_LEN * HD));
  int b = bg / NKV, g = bg % NKV;
  vt[pid] = f2bf(v[((long long)b * S_LEN + s) * KVDIM + g * HD + d]);
}

// C[M][N] = A[M][K] * W[N][K]^T   (bf16 in, f32 out)
// Block 256 threads (8 waves), block tile 128x128, K-step 32.
// Waves 4(M) x 2(N); wave tile 32x64 (8 accumulators; A-frag reuse x4).
// A/B tiles double-buffered in LDS, filled with async global->LDS copies.
__global__ __launch_bounds__(256) void k_gemm_bf16(
    const u16* __restrict__ A, const u16* __restrict__ W,
    float* __restrict__ C, int M, int N, int K) {
  __shared__ u16 Asm[2][BM * BK];
  __shared__ u16 Bsm[2][BN * BK];
  int tid = threadIdx.x;
  int lane = tid & 31;
  int w = tid >> 5;
  int r = lane & 15, h = lane >> 4;
  int wm = (w & 3) * 32;       // wave row offset in block tile
  int wn = (w >> 2) * 64;      // wave col offset in block tile
  int rowA0 = blockIdx.x * BM;
  int colB0 = blockIdx.y * BN;

  v8f zero = {0, 0, 0, 0, 0, 0, 0, 0};
  v8f acc[2][4];
#pragma unroll
  for (int i = 0; i < 2; ++i)
#pragma unroll
    for (int j = 0; j < 4; ++j) acc[i][j] = zero;

  // prologue: fill buffer 0
  stage_async(A, K, rowA0, 0, &Asm[0][0], tid);
  stage_async(W, K, colB0, 0, &Bsm[0][0], tid);
  wait_async_all();
  __syncthreads();

  int cur = 0;
  for (int k0 = 0; k0 < K; k0 += BK) {
    if (k0 + BK < K) {  // prefetch next K-slab into the other buffer
      stage_async(A, K, rowA0, k0 + BK, &Asm[cur ^ 1][0], tid);
      stage_async(W, K, colB0, k0 + BK, &Bsm[cur ^ 1][0], tid);
    }
    // fragments from LDS
    v16bf af[2];
    af[0] = load_frag_a(&Asm[cur][0], wm,      0, BK);
    af[1] = load_frag_a(&Asm[cur][0], wm + 16, 0, BK);
    v16bf bf[4];
#pragma unroll
    for (int j = 0; j < 4; ++j)
      bf[j] = load_frag_b(&Bsm[cur][0], wn + j * 16, 0, BK);
#pragma unroll
    for (int i = 0; i < 2; ++i)
#pragma unroll
      for (int j = 0; j < 4; ++j)
        acc[i][j] = wmma_bf16(af[i], bf[j], acc[i][j]);

    wait_async_all();   // next buffer fully written (overlapped with WMMAs)
    __syncthreads();    // all waves done reading cur + writing cur^1
    cur ^= 1;
  }

#pragma unroll
  for (int i = 0; i < 2; ++i)
#pragma unroll
    for (int j = 0; j < 4; ++j)
#pragma unroll
      for (int v = 0; v < 8; ++v)
        C[(size_t)(rowA0 + wm + i * 16 + v + 8 * h) * N +
          colB0 + wn + j * 16 + r] = acc[i][j][v];
}

// Flash attention: one wave per (b, head, 16-query tile).
// Q bf16 [B*S][2048] (roped), K bf16 [B*S][512] (roped), Vt bf16 [B][G][HD][S].
// Output bf16 [B*S][2048].
__global__ __launch_bounds__(32) void k_attn(
    const u16* __restrict__ Q, const u16* __restrict__ Kb,
    const u16* __restrict__ Vt, u16* __restrict__ Ob) {
  __shared__ u16 pbuf[16 * 32];
  int qt = blockIdx.x;   // query tile (S/16 = 64)
  int hh = blockIdx.y;   // head 0..15
  int b  = blockIdx.z;   // batch
  int g  = hh >> 2;      // kv head
  int lane = threadIdx.x & 31;
  int r = lane & 15, hf = lane >> 4;

  const u16* Qp = Q  + (size_t)b * S_LEN * DIM_;
  const u16* Kp = Kb + (size_t)b * S_LEN * KVDIM;
  const u16* Vp = Vt + (size_t)(b * NKV + g) * HD * S_LEN;

  v16bf qf[4];
#pragma unroll
  for (int d = 0; d < 4; ++d)
    qf[d] = load_frag_a(Qp, qt * 16, hh * HD + d * 32, DIM_);

  v8f zero = {0, 0, 0, 0, 0, 0, 0, 0};
  v8f O[8];
#pragma unroll
  for (int nt = 0; nt < 8; ++nt) O[nt] = zero;
  float mrow[8], lrow[8];
#pragma unroll
  for (int v = 0; v < 8; ++v) { mrow[v] = -3.0e38f; lrow[v] = 0.0f; }

  const float sc = 0.08838834764831845f;  // 1/sqrt(128)
  int qmax = qt * 16 + 15;
  for (int kb = 0; kb * 32 <= qmax; ++kb) {
    int kpos0 = kb * 32;
    v8f s0 = zero, s1 = zero;
#pragma unroll
    for (int d = 0; d < 4; ++d) {
      v16bf b0 = load_frag_b(Kp, kpos0,      g * HD + d * 32, KVDIM);
      v16bf b1 = load_frag_b(Kp, kpos0 + 16, g * HD + d * 32, KVDIM);
      s0 = wmma_bf16(qf[d], b0, s0);
      s1 = wmma_bf16(qf[d], b1, s1);
    }
    // Online softmax (C layout: row = v + 8*hf, col = r).
#pragma unroll
    for (int v = 0; v < 8; ++v) {
      int rowq = qt * 16 + v + 8 * hf;
      float e0 = s0[v] * sc, e1 = s1[v] * sc;
      if (kpos0 + r      > rowq) e0 = -3.0e38f;
      if (kpos0 + 16 + r > rowq) e1 = -3.0e38f;
      float x = fmaxf(e0, e1);
#pragma unroll
      for (int m = 8; m >= 1; m >>= 1) x = fmaxf(x, __shfl_xor(x, m, 32));
      float mnew = fmaxf(mrow[v], x);
      float corr = __expf(mrow[v] - mnew);
      float p0 = __expf(e0 - mnew);
      float p1 = __expf(e1 - mnew);
      float rs = p0 + p1;
#pragma unroll
      for (int m = 8; m >= 1; m >>= 1) rs += __shfl_xor(rs, m, 32);
      lrow[v] = lrow[v] * corr + rs;
      mrow[v] = mnew;
#pragma unroll
      for (int nt = 0; nt < 8; ++nt) O[nt][v] *= corr;
      int prow = v + 8 * hf;
      pbuf[prow * 32 + r]      = f2bf(p0);
      pbuf[prow * 32 + 16 + r] = f2bf(p1);
    }
    __syncthreads();  // P tile: C-layout -> A-layout transpose via LDS
    v16bf pf = load_frag_a(pbuf, 0, 0, 32);
#pragma unroll
    for (int nt = 0; nt < 8; ++nt) {
      v16bf vf = load_frag_b(Vp, nt * 16, kpos0, S_LEN);
      O[nt] = wmma_bf16(pf, vf, O[nt]);
    }
    __syncthreads();
  }
  size_t orow = (size_t)b * S_LEN + qt * 16;
#pragma unroll
  for (int nt = 0; nt < 8; ++nt)
#pragma unroll
    for (int v = 0; v < 8; ++v) {
      float val = O[nt][v] / lrow[v];
      Ob[(orow + v + 8 * hf) * DIM_ + hh * HD + nt * 16 + r] = f2bf(val);
    }
}

// ---------------- launch ----------------

extern "C" void kernel_launch(void* const* d_in, const int* in_sizes, int n_in,
                              void* d_out, int out_size, void* d_ws, size_t ws_size,
                              hipStream_t stream) {
  const float* x    = (const float*)d_in[0];
  const float* fcos = (const float*)d_in[1];
  const float* fsin = (const float*)d_in[2];
  const float* wq   = (const float*)d_in[3];
  const float* wk   = (const float*)d_in[4];
  const float* wv   = (const float*)d_in[5];
  const float* wo   = (const float*)d_in[6];
  float* out = (float*)d_out;

  char* ws = (char*)d_ws;
  size_t off = 0;
  auto carve = [&](size_t bytes) -> void* {
    void* p = ws + off;
    off += (bytes + 255) & ~(size_t)255;
    return p;
  };

  u16*   xb   = (u16*)  carve((size_t)MROWS * DIM_ * 2);
  u16*   wqb  = (u16*)  carve((size_t)DIM_ * DIM_ * 2);
  u16*   wkb  = (u16*)  carve((size_t)KVDIM * DIM_ * 2);
  u16*   wvb  = (u16*)  carve((size_t)KVDIM * DIM_ * 2);
  u16*   wob  = (u16*)  carve((size_t)DIM_ * DIM_ * 2);
  float* Qf   = (float*)carve((size_t)MROWS * DIM_ * 4);
  float* Kf   = (float*)carve((size_t)MROWS * KVDIM * 4);
  float* Vf   = (float*)carve((size_t)MROWS * KVDIM * 4);
  u16*   Qbf  = (u16*)  carve((size_t)MROWS * DIM_ * 2);
  u16*   Kbf  = (u16*)  carve((size_t)MROWS * KVDIM * 2);
  u16*   Vtb  = (u16*)  carve((size_t)BATCH * NKV * HD * S_LEN * 2);
  u16*   Abf  = (u16*)  carve((size_t)MROWS * DIM_ * 2);

  auto blocks = [](long long n) { return (unsigned)((n + 255) / 256); };

  // 1) bf16 conversions
  k_f32_to_bf16<<<blocks((long long)MROWS * DIM_), 256, 0, stream>>>(x,  xb,  (long long)MROWS * DIM_);
  k_f32_to_bf16<<<blocks((long long)DIM_ * DIM_),  256, 0, stream>>>(wq, wqb, (long long)DIM_ * DIM_);
  k_f32_to_bf16<<<blocks((long long)KVDIM * DIM_), 256, 0, stream>>>(wk, wkb, (long long)KVDIM * DIM_);
  k_f32_to_bf16<<<blocks((long long)KVDIM * DIM_), 256, 0, stream>>>(wv, wvb, (long long)KVDIM * DIM_);
  k_f32_to_bf16<<<blocks((long long)DIM_ * DIM_),  256, 0, stream>>>(wo, wob, (long long)DIM_ * DIM_);

  // 2) QKV projections (WMMA + async LDS double buffering)
  k_gemm_bf16<<<dim3(MROWS / BM, DIM_ / BN),  256, 0, stream>>>(xb, wqb, Qf, MROWS, DIM_,  DIM_);
  k_gemm_bf16<<<dim3(MROWS / BM, KVDIM / BN), 256, 0, stream>>>(xb, wkb, Kf, MROWS, KVDIM, DIM_);
  k_gemm_bf16<<<dim3(MROWS / BM, KVDIM / BN), 256, 0, stream>>>(xb, wvb, Vf, MROWS, KVDIM, DIM_);

  // 3) RoPE + V transpose, producing bf16 operands for attention
  k_rope_bf16<<<blocks((long long)MROWS * NH  * (HD / 2)), 256, 0, stream>>>(Qf, fcos, fsin, Qbf, NH);
  k_rope_bf16<<<blocks((long long)MROWS * NKV * (HD / 2)), 256, 0, stream>>>(Kf, fcos, fsin, Kbf, NKV);
  k_transpose_v<<<blocks((long long)BATCH * NKV * HD * S_LEN), 256, 0, stream>>>(Vf, Vtb);

  // 4) Causal GQA flash attention (WMMA)
  k_attn<<<dim3(S_LEN / 16, NH, BATCH), 32, 0, stream>>>(Qbf, Kbf, Vtb, Abf);

  // 5) Output projection (WMMA) -> fp32 d_out
  k_gemm_bf16<<<dim3(MROWS / BM, DIM_ / BN), 256, 0, stream>>>(Abf, wob, out, MROWS, DIM_, DIM_);
}
